// PsychProfileNetwork_10041633538377
// MI455X (gfx1250) — matrix operand order repaired and checked by reading, not compiled, not run on hardware
//
#include <hip/hip_runtime.h>

typedef __attribute__((ext_vector_type(16))) _Float16 v16h;
typedef __attribute__((ext_vector_type(8)))  _Float16 v8h;
typedef __attribute__((ext_vector_type(8)))  float    v8f;

// ---------------------------------------------------------------------------
// RNG: hash -> Box-Muller normal (reference uses jax PRNG; statistically
// equivalent noise, cannot be bit-matched).
// ---------------------------------------------------------------------------
__device__ __forceinline__ unsigned hash32(unsigned x) {
  x ^= x >> 16; x *= 0x7feb352dU;
  x ^= x >> 15; x *= 0x846ca68bU;
  x ^= x >> 16;
  return x;
}

__device__ __forceinline__ float gauss_noise(unsigned seed, unsigned t, unsigned idx) {
  unsigned h1 = hash32(seed ^ (idx * 0x9E3779B9u) ^ (t * 0x85EBCA6Bu));
  unsigned h2 = hash32(h1 + 0x9E3779B9u);
  float u1 = (float)(h1 >> 8) * (1.0f / 16777216.0f) + 1.0e-7f;
  float u2 = (float)(h2 >> 8) * (1.0f / 16777216.0f);
  float r  = __builtin_sqrtf(-2.0f * __logf(u1));
  return r * __cosf(6.28318530718f * u2);
}

// ---------------------------------------------------------------------------
// f32 -> f16 weight conversion (spikes are 0/1 so the f16 A operand is exact;
// weights ~N(0, 1/sqrt(K)) convert with negligible loss).
// ---------------------------------------------------------------------------
__global__ void cvt_f32_to_f16(const float* __restrict__ in,
                               _Float16* __restrict__ out, int n) {
  int i = blockIdx.x * blockDim.x + threadIdx.x;
  if (i < n) out[i] = (_Float16)in[i];
}

// ---------------------------------------------------------------------------
// LIF scan, one thread per (b, n) neuron, sequential over T.
// Writes spikes as f16 (WMMA A-operand friendly).
// ---------------------------------------------------------------------------
__global__ void lif_scan_f16(const float* __restrict__ I,
                             _Float16* __restrict__ spk,
                             int T, int stride, float V_th, float inv_tau,
                             float noise_std, int adaptive, float recovery,
                             unsigned seed) {
  int idx = blockIdx.x * blockDim.x + threadIdx.x;
  float V = 0.0f, th = V_th;
  for (int t = 0; t < T; ++t) {
    float nz  = gauss_noise(seed, (unsigned)t, (unsigned)idx) * noise_std;
    float i_t = I[(size_t)t * stride + idx];
    V += (i_t - V) * inv_tau + nz;
    float s = (V >= th) ? 1.0f : 0.0f;
    spk[(size_t)t * stride + idx] = (_Float16)s;
    V *= (1.0f - s);
    if (adaptive) th += recovery * (V_th - th) + 0.05f * s;
  }
}

// Final (striatum) LIF scan: writes f32 spikes and pre-reset voltages to d_out.
__global__ void lif_scan_out(const float* __restrict__ I,
                             float* __restrict__ spk_out,
                             float* __restrict__ volt_out,
                             int T, int stride, float V_th, float inv_tau,
                             float noise_std, int adaptive, float recovery,
                             unsigned seed) {
  int idx = blockIdx.x * blockDim.x + threadIdx.x;
  float V = 0.0f, th = V_th;
  for (int t = 0; t < T; ++t) {
    float nz  = gauss_noise(seed, (unsigned)t, (unsigned)idx) * noise_std;
    float i_t = I[(size_t)t * stride + idx];
    V += (i_t - V) * inv_tau + nz;
    size_t ofs = (size_t)t * stride + idx;
    volt_out[ofs] = V;                    // scan records V pre-reset
    float s = (V >= th) ? 1.0f : 0.0f;
    spk_out[ofs] = s;
    V *= (1.0f - s);
    if (adaptive) th += recovery * (V_th - th) + 0.05f * s;
  }
}

// ---------------------------------------------------------------------------
// WMMA GEMM: Cout[m, n] = sum_k A[m, k] * W[n, k]  (+ bias[n]) (* neuromod)
// One wave per 16x64 output strip: the A fragment is loaded once per K-step
// and reused across 4 v_wmma_f32_16x16x32_f16 (4 N-tiles), cutting the
// load:wmma ratio from 2.0 to 1.25 b128 per WMMA and A-side L2 traffic 4x.
// Grids are exact (M mult of 16, N mult of 64) so EXEC is all-1s as WMMA
// requires.
// ---------------------------------------------------------------------------
__global__ void wmma_gemm4(const _Float16* __restrict__ A,
                           const _Float16* __restrict__ W,
                           const float* __restrict__ bias,
                           const float* __restrict__ reward,   // (B, N) or null
                           float* __restrict__ Cout,
                           int M, int N, int K, int Bmask) {
  int wave = (int)((blockIdx.x * blockDim.x + threadIdx.x) >> 5);
  int lane = (int)(threadIdx.x & 31);
  int groups_n = N >> 6;                 // groups of 4 N-tiles (64 cols)
  int tm = wave / groups_n;
  int g  = wave % groups_n;
  int half = lane >> 4;                  // which half-wave
  int l    = lane & 15;

  // A layout (contiguous k order used here, same as B): lane<16 -> K [0..15];
  //                            lane>=16 -> K [16..31]; row M = l.
  const _Float16* arow = A + (size_t)(tm * 16 + l) * K + half * 16;
  // B layout (contiguous k order used here, same as A): lanes 0-15 hold K=0..15, lanes 16-31 K=16..31;
  // column N = l. Since y = x @ W^T, W row n is exactly B column n over K.
  const _Float16* brow = W + (size_t)(g * 64 + l) * K + half * 16;
  const size_t bstep = (size_t)16 * K;   // next N-tile (16 W rows)

  v8f acc0 = {}, acc1 = {}, acc2 = {}, acc3 = {};
  for (int k0 = 0; k0 < K; k0 += 32) {
    v8h alo = *(const v8h*)(arow + k0);          // 16B aligned
    v8h ahi = *(const v8h*)(arow + k0 + 8);
    v16h a;
#pragma unroll
    for (int i = 0; i < 8; ++i) { a[i] = alo[i]; a[i + 8] = ahi[i]; }

    v16h b0 = *(const v16h*)(brow + k0);              // 32B aligned
    v16h b1 = *(const v16h*)(brow + bstep     + k0);
    v16h b2 = *(const v16h*)(brow + bstep * 2 + k0);
    v16h b3 = *(const v16h*)(brow + bstep * 3 + k0);

    acc0 = __builtin_amdgcn_wmma_f32_16x16x32_f16(false, a, false, b0, (short)0, acc0, false, false);
    acc1 = __builtin_amdgcn_wmma_f32_16x16x32_f16(false, a, false, b1, (short)0, acc1, false, false);
    acc2 = __builtin_amdgcn_wmma_f32_16x16x32_f16(false, a, false, b2, (short)0, acc2, false, false);
    acc3 = __builtin_amdgcn_wmma_f32_16x16x32_f16(false, a, false, b3, (short)0, acc3, false, false);
  }

  // C/D layout: VGPR r, lane -> (M = r + 8*half, N = l)
  v8f accs[4] = {acc0, acc1, acc2, acc3};
#pragma unroll
  for (int j = 0; j < 4; ++j) {
    int n = g * 64 + j * 16 + l;
    float bval = bias ? bias[n] : 0.0f;
#pragma unroll
    for (int r = 0; r < 8; ++r) {
      int m = tm * 16 + r + half * 8;
      float v = accs[j][r] + bval;
      if (reward) {  // neuromod: sigmoid(3 * reward[b, n]), b = m % B (B=2^k)
        float rw = reward[(size_t)(m & (Bmask - 1)) * N + n];
        v *= 1.0f / (1.0f + __expf(-3.0f * rw));
      }
      Cout[(size_t)m * N + n] = v;
    }
  }
}

// ---------------------------------------------------------------------------
// Launch: pfc LIF -> GEMM1(+bias) -> amyg LIF -> GEMM2(*neuromod) -> striatum
// LIF -> d_out (spikes, voltages).
// Workspace layout (bytes), with dead-buffer reuse (peak ~152 MB):
//   [0,       33.5M)  pfc spikes f16     (dead after GEMM1)
//   [33.5M,   67M)    pfc activ  f32     (dead after amyg LIF)
//   [0,       134M)   striatum input f32 (overlays the two above)
//   [134M,    151M)   amyg spikes f16
//   [151M,    ...)    f16 weight copies
// ---------------------------------------------------------------------------
extern "C" void kernel_launch(void* const* d_in, const int* in_sizes, int n_in,
                              void* d_out, int out_size, void* d_ws, size_t ws_size,
                              hipStream_t stream) {
  const float* sensory = (const float*)d_in[0];   // (128, 256, 512)
  const float* reward  = (const float*)d_in[1];   // (256, 1024)
  const float* W1f     = (const float*)d_in[2];   // (256, 512)
  const float* b1      = (const float*)d_in[3];   // (256,)
  const float* W2f     = (const float*)d_in[4];   // (1024, 256)

  const int T = 128, B = 256, NP = 512, NA = 256, NS = 1024;
  const int MTB = T * B;                          // 32768 rows for the GEMMs

  char* ws = (char*)d_ws;
  const size_t OFF_A  = 0;                        // pfc spikes f16: 33,554,432
  const size_t OFF_B  = 33554432;                 // pfc activ f32: 33,554,432
  const size_t OFF_D  = 0;                        // striatum in f32: 134,217,728
  const size_t OFF_C  = 134217728;                // amyg spikes f16: 16,777,216
  const size_t OFF_W1 = OFF_C + 16777216;         // 256KB
  const size_t OFF_W2 = OFF_W1 + 262144;          // 512KB

  _Float16* pfc_spk  = (_Float16*)(ws + OFF_A);
  float*    pfc_act  = (float*)   (ws + OFF_B);
  _Float16* amyg_spk = (_Float16*)(ws + OFF_C);
  float*    str_in   = (float*)   (ws + OFF_D);
  _Float16* W1h      = (_Float16*)(ws + OFF_W1);
  _Float16* W2h      = (_Float16*)(ws + OFF_W2);

  cvt_f32_to_f16<<<(NA * NP) / 256, 256, 0, stream>>>(W1f, W1h, NA * NP);
  cvt_f32_to_f16<<<(NS * NA) / 256, 256, 0, stream>>>(W2f, W2h, NS * NA);

  // PFC: V_th=1.2, tau=30, noise 0.1, adaptive rec=0.1
  lif_scan_f16<<<(B * NP) / 256, 256, 0, stream>>>(
      sensory, pfc_spk, T, B * NP, 1.2f, 1.0f / 30.0f, 0.1f, 1, 0.1f, 0x1234u);

  // GEMM1: (32768, 512) x (512, 256) + bias -> pfc_act
  {
    int waves = (MTB / 16) * (NA / 64);           // 8192 waves (16x64 strips)
    wmma_gemm4<<<waves * 32 / 256, 256, 0, stream>>>(
        pfc_spk, W1h, b1, nullptr, pfc_act, MTB, NA, NP, B);
  }

  // Amygdala: V_th=1.0, tau=20, noise 0.3, non-adaptive
  lif_scan_f16<<<(B * NA) / 256, 256, 0, stream>>>(
      pfc_act, amyg_spk, T, B * NA, 1.0f, 1.0f / 20.0f, 0.3f, 0, 0.1f, 0x5678u);

  // GEMM2: (32768, 256) x (256, 1024), fused neuromod sigmoid(3*reward)
  {
    int waves = (MTB / 16) * (NS / 64);           // 32768 waves
    wmma_gemm4<<<waves * 32 / 256, 256, 0, stream>>>(
        amyg_spk, W2h, nullptr, reward, str_in, MTB, NS, NA, B);
  }

  // Striatum: V_th=1.0, tau=20, noise 0.1, adaptive rec=0.15 -> d_out
  float* out_spk = (float*)d_out;
  float* out_v   = out_spk + (size_t)T * B * NS;
  lif_scan_out<<<(B * NS) / 256, 256, 0, stream>>>(
      str_in, out_spk, out_v, T, B * NS, 1.0f, 1.0f / 20.0f, 0.1f, 1, 0.15f, 0x9abcu);
}
